// GAT_4346506904233
// MI455X (gfx1250) — matrix operand order repaired and checked by reading, not compiled, not run on hardware
//
#include <hip/hip_runtime.h>
#include <hip/hip_bf16.h>

#define NNODES 100000
#define DIM 128
#define NEG_SLOPE 0.2f
#define BN_EPS 1e-5f

typedef __attribute__((ext_vector_type(16))) _Float16 v16h;
typedef __attribute__((ext_vector_type(8)))  _Float16 v8h;
typedef __attribute__((ext_vector_type(4)))  _Float16 v4h;
typedef __attribute__((ext_vector_type(8)))  float    v8f;
typedef __attribute__((ext_vector_type(4)))  float    v4f;

// ---------- float <-> order-preserving uint (for atomic float max) ----------
__device__ __forceinline__ unsigned f2ord(float f) {
    unsigned u = __float_as_uint(f);
    return (u & 0x80000000u) ? ~u : (u | 0x80000000u);
}
__device__ __forceinline__ float ord2f(unsigned u) {
    return (u & 0x80000000u) ? __uint_as_float(u & 0x7fffffffu) : __uint_as_float(~u);
}
#define ORD_NEG_INF 0x007FFFFFu  // f2ord(-inf)

// ---------- f32 -> f16 conversion (vectorized by 4) ----------
__global__ __launch_bounds__(256) void k_cvt(const float* __restrict__ in,
                                             _Float16* __restrict__ out, int n4) {
    int i = blockIdx.x * 256 + threadIdx.x;
    if (i >= n4) return;
    v4f v = ((const v4f*)in)[i];
    v4h h;
    h.x = (_Float16)v.x; h.y = (_Float16)v.y; h.z = (_Float16)v.z; h.w = (_Float16)v.w;
    ((v4h*)out)[i] = h;
}

// ---------- W [dout x din] f32 -> Wt [din x dout] f16 ----------
__global__ __launch_bounds__(256) void k_wt(const float* __restrict__ W,
                                            _Float16* __restrict__ Wt) {
    int i = blockIdx.x * 256 + threadIdx.x;   // 64 blocks * 256 = 16384
    int o = i >> 7, k = i & (DIM - 1);
    Wt[k * DIM + o] = (_Float16)W[o * DIM + k];
}

// ---------- WMMA GEMM: H[N,128] = X_f16[N,128] @ Wt_f16[128,128] ----------
// Block = 8 waves; wave w computes tile (blockIdx.x*16 rows, w*16 cols).
// A fragment (16x32 f16, ISA 7.12.2): lane M = lane%16, hi = lane/16;
//   halves 0..7  = K = kk + hi*8 .. +7       (one b128 load)
//   halves 8..15 = K = kk + 16 + hi*8 .. +7  (one b128 load)
// B fragment (32x16 f16): lane holds column n0 + lane%16;
//   halves in the same K order as A  (16 strided 16-bit loads).
__global__ __launch_bounds__(256) void k_gemm(const _Float16* __restrict__ X,
                                              const _Float16* __restrict__ Wt,
                                              float* __restrict__ H) {
    const int lane = threadIdx.x & 31;
    const int wave = threadIdx.x >> 5;
    const int m0 = blockIdx.x * 16;
    const int n0 = wave * 16;
    const int M  = lane & 15;
    const int hi = lane >> 4;
    const int bK = (lane & 15) + (hi << 4);

    union Frag { v16h v; v8h h[2]; };
    v8f c = {};
    const _Float16* arow = X + (size_t)(m0 + M) * DIM;

#pragma unroll
    for (int kk = 0; kk < DIM; kk += 32) {
        Frag a, b;
        a.h[0] = *(const v8h*)(arow + kk + hi * 8);
        a.h[1] = *(const v8h*)(arow + kk + 16 + hi * 8);
        const _Float16* brow = Wt + (size_t)(kk + hi * 8) * DIM + n0 + M;
#pragma unroll
        for (int j = 0; j < 8; j++) {
            b.v[j]     = brow[(size_t)j * DIM];
            b.v[8 + j] = brow[(size_t)(16 + j) * DIM];
        }
        c = __builtin_amdgcn_wmma_f32_16x16x32_f16(false, a.v, false, b.v,
                                                   (short)0, c, false, false);
    }
    // C/D layout: lane l -> N = l%16; vgpr r -> M = r + 8*(l/16)
    float* hp = H + (size_t)(m0 + hi * 8) * DIM + n0 + (lane & 15);
#pragma unroll
    for (int r = 0; r < 8; r++) hp[(size_t)r * DIM] = c[r];
}

// ---------- alpha_s / alpha_d: per-node 128-dot, one wave per node ----------
__global__ __launch_bounds__(256) void k_alpha(const float* __restrict__ H,
                                               const float* __restrict__ asrc,
                                               const float* __restrict__ adst,
                                               float* __restrict__ As,
                                               float* __restrict__ Ad) {
    int lane = threadIdx.x & 31;
    int node = blockIdx.x * 8 + (threadIdx.x >> 5);
    if (node >= NNODES) return;
    v4f h  = ((const v4f*)(H + (size_t)node * DIM))[lane];
    v4f s4 = ((const v4f*)asrc)[lane];
    v4f d4 = ((const v4f*)adst)[lane];
    float s = h.x * s4.x + h.y * s4.y + h.z * s4.z + h.w * s4.w;
    float d = h.x * d4.x + h.y * d4.y + h.z * d4.z + h.w * d4.w;
#pragma unroll
    for (int o = 16; o > 0; o >>= 1) {
        s += __shfl_xor(s, o, 32);
        d += __shfl_xor(d, o, 32);
    }
    if (lane == 0) { As[node] = s; Ad[node] = d; }
}

// ---------- per-layer init: emax=-inf, denom=0, agg=0 ----------
__global__ __launch_bounds__(256) void k_init(unsigned* __restrict__ emax,
                                              float* __restrict__ denom,
                                              float* __restrict__ agg4 /* as v4f */) {
    int i = blockIdx.x * 256 + threadIdx.x;   // 12500*256 = 3.2M = N*DIM/4
    ((v4f*)agg4)[i] = (v4f){0.f, 0.f, 0.f, 0.f};
    if (i < NNODES) { emax[i] = ORD_NEG_INF; denom[i] = 0.f; }
}

__device__ __forceinline__ void edge_sd(const int* __restrict__ ei, int E, int e,
                                        int& s, int& d) {
    if (e < E) { s = ei[e]; d = ei[E + e]; }
    else       { s = d = e - E; }          // self-loop
}

// ---------- edge pass 1: leaky_relu + segment max ----------
__global__ __launch_bounds__(256) void k_edge1(const int* __restrict__ ei, int E, int Etot,
                                               const float* __restrict__ As,
                                               const float* __restrict__ Ad,
                                               float* __restrict__ ebuf,
                                               unsigned* __restrict__ emax) {
    int e = blockIdx.x * 256 + threadIdx.x;
    if (e >= Etot) return;
    int s, d; edge_sd(ei, E, e, s, d);
    float v = As[s] + Ad[d];
    float lr = v > 0.f ? v : NEG_SLOPE * v;
    ebuf[e] = lr;
    atomicMax(emax + d, f2ord(lr));
}

// ---------- edge pass 2: exp + segment sum ----------
__global__ __launch_bounds__(256) void k_edge2(const int* __restrict__ ei, int E, int Etot,
                                               float* __restrict__ ebuf,
                                               const unsigned* __restrict__ emax,
                                               float* __restrict__ denom) {
    int e = blockIdx.x * 256 + threadIdx.x;
    if (e >= Etot) return;
    int s, d; edge_sd(ei, E, e, s, d);
    (void)s;
    float ex = expf(ebuf[e] - ord2f(emax[d]));
    ebuf[e] = ex;
    atomicAdd(denom + d, ex);
}

// ---------- edge pass 3: agg[dst] += alpha * h[src]; one wave per edge ----------
__global__ __launch_bounds__(256) void k_agg(const int* __restrict__ ei, int E, int Etot,
                                             const float* __restrict__ ebuf,
                                             const float* __restrict__ denom,
                                             const float* __restrict__ H,
                                             float* __restrict__ agg) {
    int lane = threadIdx.x & 31;
    int e = blockIdx.x * 8 + (threadIdx.x >> 5);
    if (e >= Etot) return;
    int s, d; edge_sd(ei, E, e, s, d);
    float alpha = ebuf[e] / denom[d];
    v4f hv = ((const v4f*)(H + (size_t)s * DIM))[lane];
    float* o = agg + (size_t)d * DIM + lane * 4;
    atomicAdd(o + 0, alpha * hv.x);
    atomicAdd(o + 1, alpha * hv.y);
    atomicAdd(o + 2, alpha * hv.z);
    atomicAdd(o + 3, alpha * hv.w);
}

// ---------- batch norm ----------
__global__ void k_bnzero(float* cs, float* cq) {
    int i = threadIdx.x;
    if (i < DIM) { cs[i] = 0.f; cq[i] = 0.f; }
}

__global__ __launch_bounds__(256) void k_bnstats(const float* __restrict__ agg,
                                                 float* __restrict__ cs,
                                                 float* __restrict__ cq) {
    int c = threadIdx.x & (DIM - 1);
    int half = threadIdx.x >> 7;
    float s = 0.f, q = 0.f;
    for (int r = blockIdx.x * 2 + half; r < NNODES; r += gridDim.x * 2) {
        float v = agg[(size_t)r * DIM + c];
        s += v; q += v * v;
    }
    atomicAdd(cs + c, s);
    atomicAdd(cq + c, q);
}

// note: bias b cancels inside batch-norm (pure mean shift), so it is omitted.
__global__ __launch_bounds__(256) void k_bnapply(const float* __restrict__ agg,
                                                 const float* __restrict__ cs,
                                                 const float* __restrict__ cq,
                                                 const float* __restrict__ gamma,
                                                 const float* __restrict__ beta,
                                                 _Float16* __restrict__ out) {
    int i = blockIdx.x * 256 + threadIdx.x;   // 50000*256 = N*DIM
    int c = i & (DIM - 1);
    float mu  = cs[c] * (1.0f / NNODES);
    float var = cq[c] * (1.0f / NNODES) - mu * mu;
    float y = gamma[c] * (agg[i] - mu) * rsqrtf(var + BN_EPS) + beta[c];
    out[i] = (_Float16)(y > 0.f ? y : 0.f);
}

// ---------- final layer: out = agg + b ----------
__global__ __launch_bounds__(256) void k_out(const float* __restrict__ agg,
                                             const float* __restrict__ b,
                                             float* __restrict__ out) {
    int i = blockIdx.x * 256 + threadIdx.x;
    out[i] = agg[i] + b[i & (DIM - 1)];
}

extern "C" void kernel_launch(void* const* d_in, const int* in_sizes, int n_in,
                              void* d_out, int out_size, void* d_ws, size_t ws_size,
                              hipStream_t stream) {
    const float* x  = (const float*)d_in[0];
    const int*   ei = (const int*)d_in[1];
    const float* W[3]   = {(const float*)d_in[2],  (const float*)d_in[6],  (const float*)d_in[10]};
    const float* asr[3] = {(const float*)d_in[3],  (const float*)d_in[7],  (const float*)d_in[11]};
    const float* ads[3] = {(const float*)d_in[4],  (const float*)d_in[8],  (const float*)d_in[12]};
    const float* bb[3]  = {(const float*)d_in[5],  (const float*)d_in[9],  (const float*)d_in[13]};
    const float* gamma[2] = {(const float*)d_in[14], (const float*)d_in[16]};
    const float* beta[2]  = {(const float*)d_in[15], (const float*)d_in[17]};

    const int E = in_sizes[1] / 2;
    const int Etot = E + NNODES;

    // workspace carve-out
    char* base = (char*)d_ws;
    size_t off = 0;
    auto carve = [&](size_t bytes) -> char* {
        char* p = base + off;
        off = (off + bytes + 255) & ~(size_t)255;
        return p;
    };
    _Float16* inF16 = (_Float16*)carve((size_t)NNODES * DIM * 2);
    _Float16* wt    = (_Float16*)carve((size_t)DIM * DIM * 2);
    float* H    = (float*)carve((size_t)NNODES * DIM * 4);
    float* AGG  = (float*)carve((size_t)NNODES * DIM * 4);
    float* As   = (float*)carve((size_t)NNODES * 4);
    float* Ad   = (float*)carve((size_t)NNODES * 4);
    unsigned* EmaxU = (unsigned*)carve((size_t)NNODES * 4);
    float* Den  = (float*)carve((size_t)NNODES * 4);
    float* Ebuf = (float*)carve((size_t)Etot * 4);
    float* Cs   = (float*)carve(DIM * 4);
    float* Cq   = (float*)carve(DIM * 4);

    const int gEdge = (Etot + 255) / 256;
    const int gAgg  = (Etot + 7) / 8;

    k_cvt<<<(NNODES * DIM / 4 + 255) / 256, 256, 0, stream>>>(x, inF16, NNODES * DIM / 4);

    for (int l = 0; l < 3; l++) {
        k_wt<<<64, 256, 0, stream>>>(W[l], wt);
        k_gemm<<<NNODES / 16, 256, 0, stream>>>(inF16, wt, H);
        k_alpha<<<(NNODES + 7) / 8, 256, 0, stream>>>(H, asr[l], ads[l], As, Ad);
        k_init<<<NNODES * DIM / 4 / 256, 256, 0, stream>>>(EmaxU, Den, AGG);
        k_edge1<<<gEdge, 256, 0, stream>>>(ei, E, Etot, As, Ad, Ebuf, EmaxU);
        k_edge2<<<gEdge, 256, 0, stream>>>(ei, E, Etot, Ebuf, EmaxU, Den);
        k_agg<<<gAgg, 256, 0, stream>>>(ei, E, Etot, Ebuf, Den, H, AGG);
        if (l < 2) {
            k_bnzero<<<1, 128, 0, stream>>>(Cs, Cq);
            k_bnstats<<<1024, 256, 0, stream>>>(AGG, Cs, Cq);
            k_bnapply<<<NNODES * DIM / 256, 256, 0, stream>>>(AGG, Cs, Cq,
                                                              gamma[l], beta[l], inF16);
        } else {
            k_out<<<NNODES * DIM / 256, 256, 0, stream>>>(AGG, bb[2], (float*)d_out);
        }
    }
}
